// Seq2seq_42271068127633
// MI455X (gfx1250) — matrix operand order repaired and checked by reading, not compiled, not run on hardware
//
#include <hip/hip_runtime.h>

#define F_ 128
#define H_ 256
#define ISEQ 96
#define PSEQ 48
#define BATCH 4096

typedef __attribute__((ext_vector_type(16))) __bf16 v16bf;
typedef __attribute__((ext_vector_type(8)))  float  v8f;

union Frag {
    v16bf v;
    uint4 u[2];
    unsigned short s[16];
};

__device__ __forceinline__ unsigned short f2bf(float f) {
    unsigned u = __builtin_bit_cast(unsigned, f);
    u += 0x7FFFu + ((u >> 16) & 1u);           // round-to-nearest-even
    return (unsigned short)(u >> 16);
}
__device__ __forceinline__ float bf2f(unsigned short s) {
    unsigned u = ((unsigned)s) << 16;
    return __builtin_bit_cast(float, u);
}
__device__ __forceinline__ float sigmoid_(float x) {
    return 1.0f / (1.0f + __expf(-x));
}

// ---- WMMA wrapper: D = A(16x32 bf16) * B(32x16 bf16) + C(16x16 f32) ----
__device__ __forceinline__ v8f wmma_bf16(v16bf a, v16bf b, v8f c) {
    return __builtin_amdgcn_wmma_f32_16x16x32_bf16(
        /*neg_a=*/false, a, /*neg_b=*/false, b,
        /*c_mod=*/(short)0, c, /*reuse_a=*/false, /*reuse_b=*/false);
}

// A fragment (16x32, bf16 source, row-major, element stride `stride`).
// 16-bit A layout: lane holds M=lane&15, K = {8hi+0..7} U {16+8hi+0..7}.
__device__ __forceinline__ v16bf loadA_bf16(const unsigned short* __restrict__ base,
                                            int stride, int row0, int k0, int lane) {
    int r = lane & 15, hi = lane >> 4;
    const unsigned short* p = base + (size_t)(row0 + r) * stride + k0 + 8 * hi;
    Frag f;
    f.u[0] = *reinterpret_cast<const uint4*>(p);        // K = k0+8hi .. +7
    f.u[1] = *reinterpret_cast<const uint4*>(p + 16);   // K = k0+16+8hi .. +7
    return f.v;
}

// A fragment from fp32 source (fallback path only; converts on the fly).
__device__ __forceinline__ v16bf loadA_f32(const float* __restrict__ base,
                                           int stride, int row0, int k0, int lane) {
    int r = lane & 15, hi = lane >> 4;
    const float* p = base + (size_t)(row0 + r) * stride + k0 + 8 * hi;
    float4 a0 = *reinterpret_cast<const float4*>(p);
    float4 a1 = *reinterpret_cast<const float4*>(p + 4);
    float4 b0 = *reinterpret_cast<const float4*>(p + 16);
    float4 b1 = *reinterpret_cast<const float4*>(p + 20);
    Frag f;
    f.s[0]  = f2bf(a0.x); f.s[1]  = f2bf(a0.y); f.s[2]  = f2bf(a0.z); f.s[3]  = f2bf(a0.w);
    f.s[4]  = f2bf(a1.x); f.s[5]  = f2bf(a1.y); f.s[6]  = f2bf(a1.z); f.s[7]  = f2bf(a1.w);
    f.s[8]  = f2bf(b0.x); f.s[9]  = f2bf(b0.y); f.s[10] = f2bf(b0.z); f.s[11] = f2bf(b0.w);
    f.s[12] = f2bf(b1.x); f.s[13] = f2bf(b1.y); f.s[14] = f2bf(b1.z); f.s[15] = f2bf(b1.w);
    return f.v;
}

// B fragment (32x16) from weight matrix W stored row-major (N x K) in bf16:
// B[k][n] = W[n][k]. Lane holds N=lane&15, K = {8hi+0..7} U {16+8hi+0..7}.
__device__ __forceinline__ v16bf loadB(const unsigned short* __restrict__ W,
                                       int K, int n0, int k0, int lane) {
    int n = lane & 15, hi = lane >> 4;
    const unsigned short* p = W + (size_t)(n0 + n) * K + k0 + 8 * hi;
    Frag f;
    f.u[0] = *reinterpret_cast<const uint4*>(p);
    f.u[1] = *reinterpret_cast<const uint4*>(p + 16);
    return f.v;
}

// ---------------- vectorized fp32 -> bf16 conversion (one-time passes) ----------------
__global__ void cvt4_f32_bf16_kernel(const float* __restrict__ src,
                                     unsigned short* __restrict__ dst, int n4) {
    int i = blockIdx.x * blockDim.x + threadIdx.x;
    if (i < n4) {
        float4 f = reinterpret_cast<const float4*>(src)[i];
        ushort4 o;
        o.x = f2bf(f.x); o.y = f2bf(f.y); o.z = f2bf(f.z); o.w = f2bf(f.w);
        reinterpret_cast<ushort4*>(dst)[i] = o;
    }
}

// Gather x_last = z_in[:, 95, :] and convert to bf16 (contiguous B x 128).
__global__ void gather_xlast_kernel(const float* __restrict__ z_in,
                                    unsigned short* __restrict__ xlast, int n4) {
    int i = blockIdx.x * blockDim.x + threadIdx.x;
    if (i < n4) {
        int b = i >> 5;            // 32 float4 per 128-wide row
        int c = i & 31;
        float4 f = *reinterpret_cast<const float4*>(
            z_in + ((size_t)b * ISEQ + 95) * F_ + c * 4);
        ushort4 o;
        o.x = f2bf(f.x); o.y = f2bf(f.y); o.z = f2bf(f.z); o.w = f2bf(f.w);
        reinterpret_cast<ushort4*>(xlast)[i] = o;
    }
}

// ---------------- encoder: h_enc = GRU(x_last, h0=0) -> bf16 ----------------
// h0 = 0  =>  gh = b_hh, h = (1-z)*n.   One wave = one 16x16 tile of h_enc.
__global__ void encoder_kernel(const unsigned short* __restrict__ xlast,
                               const unsigned short* __restrict__ Wih,
                               const float* __restrict__ b_ih,
                               const float* __restrict__ b_hh,
                               unsigned short* __restrict__ h_enc) {
    int lane = threadIdx.x;
    int m0 = blockIdx.x * 16;
    int c0 = blockIdx.y * 16;
    v8f ar = {}, az = {}, an = {};
#pragma unroll
    for (int k = 0; k < F_; k += 32) {
        v16bf a = loadA_bf16(xlast, F_, m0, k, lane);
        ar = wmma_bf16(a, loadB(Wih, F_, 0 * H_ + c0, k, lane), ar);
        az = wmma_bf16(a, loadB(Wih, F_, 1 * H_ + c0, k, lane), az);
        an = wmma_bf16(a, loadB(Wih, F_, 2 * H_ + c0, k, lane), an);
    }
    int col = c0 + (lane & 15), hi = lane >> 4;
    float bir = b_ih[col], biz = b_ih[H_ + col], bin = b_ih[2 * H_ + col];
    float bhr = b_hh[col], bhz = b_hh[H_ + col], bhn = b_hh[2 * H_ + col];
#pragma unroll
    for (int r = 0; r < 8; ++r) {
        int row = m0 + r + 8 * hi;
        float rg = sigmoid_(ar[r] + bir + bhr);
        float zg = sigmoid_(az[r] + biz + bhz);
        float ng = tanhf(an[r] + bin + rg * bhn);
        h_enc[(size_t)row * H_ + col] = f2bf((1.0f - zg) * ng);
    }
}

// ---------------- fc: h0_dec = relu(h_enc @ W_fc^T + b_fc) -> bf16 ----------------
__global__ void fc_kernel(const unsigned short* __restrict__ h_enc,
                          const unsigned short* __restrict__ Wfc,
                          const float* __restrict__ b_fc,
                          unsigned short* __restrict__ h0) {
    int lane = threadIdx.x;
    int m0 = blockIdx.x * 16;
    int c0 = blockIdx.y * 16;
    v8f acc = {};
#pragma unroll
    for (int k = 0; k < H_; k += 32) {
        v16bf a = loadA_bf16(h_enc, H_, m0, k, lane);
        acc = wmma_bf16(a, loadB(Wfc, H_, c0, k, lane), acc);
    }
    int col = c0 + (lane & 15), hi = lane >> 4;
    float bias = b_fc[col];
#pragma unroll
    for (int r = 0; r < 8; ++r) {
        float v = acc[r] + bias;
        h0[(size_t)(m0 + r + 8 * hi) * F_ + col] = f2bf(v > 0.0f ? v : 0.0f);
    }
}

// ---------------- decoder GRU step, all-bf16 operands (fast path) ----------------
// One wave = one 16x16 tile of h_new; 6 gate accumulators, 24 WMMAs total.
__global__ void decoder_step_bf16_kernel(const unsigned short* __restrict__ xbase, int xstride,
                                         const unsigned short* __restrict__ h_in,
                                         const unsigned short* __restrict__ Wih,
                                         const unsigned short* __restrict__ Whh,
                                         const float* __restrict__ b_ih,
                                         const float* __restrict__ b_hh,
                                         unsigned short* __restrict__ h_out,
                                         float* __restrict__ out, int t) {
    int lane = threadIdx.x;
    int m0 = blockIdx.x * 16;
    int c0 = blockIdx.y * 16;
    v8f ir = {}, iz = {}, in_ = {}, hr = {}, hz = {}, hn = {};
#pragma unroll
    for (int k = 0; k < F_; k += 32) {
        v16bf ax = loadA_bf16(xbase, xstride, m0, k, lane);
        v16bf ah = loadA_bf16(h_in, F_, m0, k, lane);
        ir  = wmma_bf16(ax, loadB(Wih, F_, 0 * F_ + c0, k, lane), ir);
        iz  = wmma_bf16(ax, loadB(Wih, F_, 1 * F_ + c0, k, lane), iz);
        in_ = wmma_bf16(ax, loadB(Wih, F_, 2 * F_ + c0, k, lane), in_);
        hr  = wmma_bf16(ah, loadB(Whh, F_, 0 * F_ + c0, k, lane), hr);
        hz  = wmma_bf16(ah, loadB(Whh, F_, 1 * F_ + c0, k, lane), hz);
        hn  = wmma_bf16(ah, loadB(Whh, F_, 2 * F_ + c0, k, lane), hn);
    }
    int col = c0 + (lane & 15), hi = lane >> 4;
    float bir = b_ih[col], biz = b_ih[F_ + col], bin = b_ih[2 * F_ + col];
    float bhr = b_hh[col], bhz = b_hh[F_ + col], bhn = b_hh[2 * F_ + col];
#pragma unroll
    for (int r = 0; r < 8; ++r) {
        int row = m0 + r + 8 * hi;
        float hp = bf2f(h_in[(size_t)row * F_ + col]);
        float rg = sigmoid_(ir[r] + bir + hr[r] + bhr);
        float zg = sigmoid_(iz[r] + biz + hz[r] + bhz);
        float ng = tanhf(in_[r] + bin + rg * (hn[r] + bhn));
        float h  = (1.0f - zg) * ng + zg * hp;
        out[(size_t)row * (PSEQ * F_) + (size_t)t * F_ + col] = h;  // (B, 48, 128)
        h_out[(size_t)row * F_ + col] = f2bf(h);
    }
}

// ---------------- decoder GRU step, fp32 x operand (fallback, small ws) ----------------
__global__ void decoder_step_f32_kernel(const float* __restrict__ xbase, int xstride,
                                        const unsigned short* __restrict__ h_in,
                                        const unsigned short* __restrict__ Wih,
                                        const unsigned short* __restrict__ Whh,
                                        const float* __restrict__ b_ih,
                                        const float* __restrict__ b_hh,
                                        unsigned short* __restrict__ h_out,
                                        float* __restrict__ out, int t) {
    int lane = threadIdx.x;
    int m0 = blockIdx.x * 16;
    int c0 = blockIdx.y * 16;
    v8f ir = {}, iz = {}, in_ = {}, hr = {}, hz = {}, hn = {};
#pragma unroll
    for (int k = 0; k < F_; k += 32) {
        v16bf ax = loadA_f32(xbase, xstride, m0, k, lane);
        v16bf ah = loadA_bf16(h_in, F_, m0, k, lane);
        ir  = wmma_bf16(ax, loadB(Wih, F_, 0 * F_ + c0, k, lane), ir);
        iz  = wmma_bf16(ax, loadB(Wih, F_, 1 * F_ + c0, k, lane), iz);
        in_ = wmma_bf16(ax, loadB(Wih, F_, 2 * F_ + c0, k, lane), in_);
        hr  = wmma_bf16(ah, loadB(Whh, F_, 0 * F_ + c0, k, lane), hr);
        hz  = wmma_bf16(ah, loadB(Whh, F_, 1 * F_ + c0, k, lane), hz);
        hn  = wmma_bf16(ah, loadB(Whh, F_, 2 * F_ + c0, k, lane), hn);
    }
    int col = c0 + (lane & 15), hi = lane >> 4;
    float bir = b_ih[col], biz = b_ih[F_ + col], bin = b_ih[2 * F_ + col];
    float bhr = b_hh[col], bhz = b_hh[F_ + col], bhn = b_hh[2 * F_ + col];
#pragma unroll
    for (int r = 0; r < 8; ++r) {
        int row = m0 + r + 8 * hi;
        float hp = bf2f(h_in[(size_t)row * F_ + col]);
        float rg = sigmoid_(ir[r] + bir + hr[r] + bhr);
        float zg = sigmoid_(iz[r] + biz + hz[r] + bhz);
        float ng = tanhf(in_[r] + bin + rg * (hn[r] + bhn));
        float h  = (1.0f - zg) * ng + zg * hp;
        out[(size_t)row * (PSEQ * F_) + (size_t)t * F_ + col] = h;
        h_out[(size_t)row * F_ + col] = f2bf(h);
    }
}

extern "C" void kernel_launch(void* const* d_in, const int* in_sizes, int n_in,
                              void* d_out, int out_size, void* d_ws, size_t ws_size,
                              hipStream_t stream) {
    (void)in_sizes; (void)n_in; (void)out_size;
    const float* z_in     = (const float*)d_in[0];
    const float* z_tar    = (const float*)d_in[1];
    const float* W_ih_enc = (const float*)d_in[2];
    /* d_in[3] = W_hh_enc: dead (h0 == 0) */
    const float* b_ih_enc = (const float*)d_in[4];
    const float* b_hh_enc = (const float*)d_in[5];
    const float* W_fc     = (const float*)d_in[6];
    const float* b_fc     = (const float*)d_in[7];
    const float* W_ih_dec = (const float*)d_in[8];
    const float* W_hh_dec = (const float*)d_in[9];
    const float* b_ih_dec = (const float*)d_in[10];
    const float* b_hh_dec = (const float*)d_in[11];
    float* out = (float*)d_out;

    // Workspace carve-up
    char* ws = (char*)d_ws;
    size_t off = 0;
    auto alloc = [&](size_t bytes) -> void* {
        void* p = ws + off;
        off += (bytes + 255) & ~(size_t)255;
        return p;
    };
    unsigned short* wih_enc_bf = (unsigned short*)alloc((size_t)3 * H_ * F_ * 2);
    unsigned short* wfc_bf     = (unsigned short*)alloc((size_t)F_ * H_ * 2);
    unsigned short* wih_dec_bf = (unsigned short*)alloc((size_t)3 * F_ * F_ * 2);
    unsigned short* whh_dec_bf = (unsigned short*)alloc((size_t)3 * F_ * F_ * 2);
    unsigned short* henc_bf    = (unsigned short*)alloc((size_t)BATCH * H_ * 2);
    unsigned short* hbuf0      = (unsigned short*)alloc((size_t)BATCH * F_ * 2);
    unsigned short* hbuf1      = (unsigned short*)alloc((size_t)BATCH * F_ * 2);
    unsigned short* xlast_bf   = (unsigned short*)alloc((size_t)BATCH * F_ * 2);
    size_t base_bytes = off;
    unsigned short* ztar_bf    = (unsigned short*)alloc((size_t)BATCH * PSEQ * F_ * 2);
    size_t full_bytes = off;
    // ws_size is a launch-invariant: this host-side branch is deterministic.
    const bool full_path = (ws_size >= full_bytes) && (ws_size >= base_bytes);

    // 1) One-time conversions (bandwidth-bound, off the recurrent critical path)
    cvt4_f32_bf16_kernel<<<(3 * H_ * F_ / 4 + 255) / 256, 256, 0, stream>>>(
        W_ih_enc, wih_enc_bf, 3 * H_ * F_ / 4);
    cvt4_f32_bf16_kernel<<<(F_ * H_ / 4 + 255) / 256, 256, 0, stream>>>(
        W_fc, wfc_bf, F_ * H_ / 4);
    cvt4_f32_bf16_kernel<<<(3 * F_ * F_ / 4 + 255) / 256, 256, 0, stream>>>(
        W_ih_dec, wih_dec_bf, 3 * F_ * F_ / 4);
    cvt4_f32_bf16_kernel<<<(3 * F_ * F_ / 4 + 255) / 256, 256, 0, stream>>>(
        W_hh_dec, whh_dec_bf, 3 * F_ * F_ / 4);
    gather_xlast_kernel<<<(BATCH * F_ / 4 + 255) / 256, 256, 0, stream>>>(
        z_in, xlast_bf, BATCH * F_ / 4);
    if (full_path) {
        int n4 = BATCH * PSEQ * F_ / 4;
        cvt4_f32_bf16_kernel<<<(n4 + 255) / 256, 256, 0, stream>>>(z_tar, ztar_bf, n4);
    }

    // 2) Encoder cell (h0 = 0) + fc head
    encoder_kernel<<<dim3(BATCH / 16, H_ / 16), 32, 0, stream>>>(
        xlast_bf, wih_enc_bf, b_ih_enc, b_hh_enc, henc_bf);
    fc_kernel<<<dim3(BATCH / 16, F_ / 16), 32, 0, stream>>>(
        henc_bf, wfc_bf, b_fc, hbuf0);

    // 3) 48 recurrent decoder steps; stream ordering = grid-wide sync
    unsigned short* hin = hbuf0;
    unsigned short* hout = hbuf1;
    dim3 dgrid(BATCH / 16, F_ / 16);
    for (int t = 0; t < PSEQ; ++t) {
        if (full_path) {
            const unsigned short* xb = (t == 0) ? xlast_bf : (ztar_bf + (size_t)(t - 1) * F_);
            int xs = (t == 0) ? F_ : (PSEQ * F_);
            decoder_step_bf16_kernel<<<dgrid, 32, 0, stream>>>(
                xb, xs, hin, wih_dec_bf, whh_dec_bf, b_ih_dec, b_hh_dec, hout, out, t);
        } else {
            const float* xb = (t == 0) ? (z_in + 95 * F_) : (z_tar + (size_t)(t - 1) * F_);
            int xs = (t == 0) ? (ISEQ * F_) : (PSEQ * F_);
            decoder_step_f32_kernel<<<dgrid, 32, 0, stream>>>(
                xb, xs, hin, wih_dec_bf, whh_dec_bf, b_ih_dec, b_hh_dec, hout, out, t);
        }
        unsigned short* tmp = hin; hin = hout; hout = tmp;
    }
}
